// SharedQKVChainAwareAttention_34033320853805
// MI455X (gfx1250) — matrix operand order repaired and checked by reading, not compiled, not run on hardware
//
#include <hip/hip_runtime.h>
#include <hip/hip_bf16.h>

typedef _Float16 half_t;
typedef __attribute__((ext_vector_type(16))) _Float16 v16h;
typedef __attribute__((ext_vector_type(8)))  _Float16 v8h;
typedef __attribute__((ext_vector_type(8)))  float    v8f;

#define B_   16
#define S_   512
#define H_   12
#define D_   64
#define DM_  768
#define ROWS_ (B_ * S_)          // 8192
#define SCALE_ 0.125f            // 64^-0.5

// ---------------------------------------------------------------------------
// A-fragment loader: 16x32 f16 A tile, lane (l,hi) holds row l.
// Element idx 0..7  -> K = k0 + 8*hi + idx        (16-byte load)
// Element idx 8..15 -> K = k0 + 16 + 8*hi + idx-8 (16-byte load)
// Works for global and LDS pointers.
// ---------------------------------------------------------------------------
template <typename PT>
__device__ __forceinline__ v16h load_a_frag(const PT* rowbase_k0, int hi) {
    const PT* p = rowbase_k0 + 8 * hi;
    v8h lo = *(const v8h*)(p);
    v8h hh = *(const v8h*)(p + 16);
    v16h r;
#pragma unroll
    for (int i = 0; i < 8; ++i) { r[i] = lo[i]; r[8 + i] = hh[i]; }
    return r;
}

// B-fragment: 32x16 f16 B tile, lane (l,hi) holds col l, same K order as A:
// idx 0..7 -> K = k0 + 8*hi + idx, idx 8..15 -> K = k0 + 16 + 8*hi + idx-8
// (two 16-byte loads along K; requires K-major storage).
__device__ __forceinline__ v16h load_b_frag(const half_t* colbase_k0, int hi) {
    return load_a_frag(colbase_k0, hi);
}

// ---------------------------------------------------------------------------
// f32 -> f16 copy
// ---------------------------------------------------------------------------
__global__ void cvt_f32_f16_kernel(const float* __restrict__ in,
                                   half_t* __restrict__ out, int n) {
    for (int i = blockIdx.x * blockDim.x + threadIdx.x; i < n;
         i += gridDim.x * blockDim.x)
        out[i] = (half_t)in[i];
}

// W[e,f] (rows x cols, row-major) -> WT[f,e] f16
__global__ void transpose_cvt_kernel(const float* __restrict__ W,
                                     half_t* __restrict__ WT,
                                     int rows, int cols) {
    int n = rows * cols;
    for (int i = blockIdx.x * blockDim.x + threadIdx.x; i < n;
         i += gridDim.x * blockDim.x) {
        int e = i / cols, f = i % cols;
        WT[(size_t)f * rows + e] = (half_t)W[i];
    }
}

// ---------------------------------------------------------------------------
// TN GEMM:  C[M,N] = A[M,K] (f16 row-major) x BT[N,K] (f16 row-major, i.e. B^T)
// One wave computes a 16x64 tile: one A-fragment is reused by 4 B-fragments /
// 4 f32 accumulators per 32-wide k-step (4 v_wmma per A load -> 2x the
// register-level arithmetic intensity of a 16x16-per-wave design).
// ---------------------------------------------------------------------------
template <bool F16OUT>
__global__ void __launch_bounds__(128)
gemm_tn_kernel(const half_t* __restrict__ A,
               const half_t* __restrict__ BT,
               void* __restrict__ Cout,
               int M, int N, int K) {
    int wave = threadIdx.x >> 5;
    int lane = threadIdx.x & 31;
    int l = lane & 15, hi = lane >> 4;

    int mt = M >> 4;                  // # of 16-row tiles
    int ng = N >> 6;                  // # of 64-col groups
    int tile = blockIdx.x * 4 + wave;
    if (tile >= mt * ng) return;
    int tm = tile % mt, tg = tile / mt;

    const half_t* arow = A + (size_t)(tm * 16 + l) * K;
    const half_t* bcol = BT + (size_t)(tg * 64 + l) * K;   // j-th subtile: +j*16*K

    v8f acc0 = {}, acc1 = {}, acc2 = {}, acc3 = {};
    for (int k0 = 0; k0 < K; k0 += 32) {
        __builtin_prefetch(arow + k0 + 128, 0, 0);   // global_prefetch_b8
        v16h af = load_a_frag(arow + k0, hi);
        v16h b0 = load_b_frag(bcol + k0, hi);
        v16h b1 = load_b_frag(bcol + (size_t)16 * K + k0, hi);
        v16h b2 = load_b_frag(bcol + (size_t)32 * K + k0, hi);
        v16h b3 = load_b_frag(bcol + (size_t)48 * K + k0, hi);
        acc0 = __builtin_amdgcn_wmma_f32_16x16x32_f16(false, af, false, b0,
                                                      (short)0, acc0, false, false);
        acc1 = __builtin_amdgcn_wmma_f32_16x16x32_f16(false, af, false, b1,
                                                      (short)0, acc1, false, false);
        acc2 = __builtin_amdgcn_wmma_f32_16x16x32_f16(false, af, false, b2,
                                                      (short)0, acc2, false, false);
        acc3 = __builtin_amdgcn_wmma_f32_16x16x32_f16(false, af, false, b3,
                                                      (short)0, acc3, false, false);
    }

    v8f accs[4] = {acc0, acc1, acc2, acc3};
#pragma unroll
    for (int j = 0; j < 4; ++j) {
#pragma unroll
        for (int r = 0; r < 8; ++r) {
            size_t row = (size_t)tm * 16 + r + 8 * hi;
            size_t col = (size_t)tg * 64 + j * 16 + l;
            if (F16OUT)
                ((half_t*)Cout)[row * N + col] = (half_t)accs[j][r];
            else
                ((float*)Cout)[row * N + col] = accs[j][r];
        }
    }
}

// ---------------------------------------------------------------------------
// RoPE on q,k (f16 [b,s,h,64]) -> qr,kr (same layout); also V -> VT [b,h,64,512]
// ---------------------------------------------------------------------------
__global__ void rope_vt_kernel(const half_t* __restrict__ qh,
                               const half_t* __restrict__ kh,
                               const half_t* __restrict__ vh,
                               half_t* __restrict__ qr,
                               half_t* __restrict__ kr,
                               half_t* __restrict__ vT) {
    int total = ROWS_ * H_ * 32;
    for (int idx = blockIdx.x * blockDim.x + threadIdx.x; idx < total;
         idx += gridDim.x * blockDim.x) {
        int i = idx & 31;
        int h = (idx >> 5) % H_;
        int row = idx / (32 * H_);
        int s = row & (S_ - 1);
        int b = row >> 9;

        float ang = (float)s * __expf(-((float)i / 32.0f) * __logf(10000.0f));
        float sn, cs;
        __sincosf(ang, &sn, &cs);

        size_t base = (size_t)row * (H_ * D_) + h * D_;
        float q1 = (float)qh[base + i], q2 = (float)qh[base + 32 + i];
        float k1 = (float)kh[base + i], k2 = (float)kh[base + 32 + i];
        qr[base + i]      = (half_t)(q1 * cs - q2 * sn);
        qr[base + 32 + i] = (half_t)(q1 * sn + q2 * cs);
        kr[base + i]      = (half_t)(k1 * cs - k2 * sn);
        kr[base + 32 + i] = (half_t)(k1 * sn + k2 * cs);

        size_t vtb = ((size_t)(b * H_ + h) * D_);
        vT[(vtb + i) * S_ + s]      = vh[base + i];
        vT[(vtb + 32 + i) * S_ + s] = vh[base + 32 + i];
    }
}

// ---------------------------------------------------------------------------
// Chain-aware attention. One block = (b, h, 16-query strip). 4 waves / block.
// Phase 1: dual score accumulators (rope'd + plain) via WMMA, per-element
//          select by chain id, +mask, f32 strip in LDS.
// Phase 2: parallel row softmax (8 threads/row, LDS tree) -> f16 P in LDS.
// Phase 3: P (LDS, ds_read A-frags) x VT (global B-frags) via WMMA.
// ---------------------------------------------------------------------------
__global__ void __launch_bounds__(128)
attn_kernel(const half_t* __restrict__ qh, const half_t* __restrict__ kh,
            const half_t* __restrict__ qr, const half_t* __restrict__ kr,
            const half_t* __restrict__ vT,
            const int* __restrict__ chain,
            const unsigned char* __restrict__ amask,
            half_t* __restrict__ aout) {
    __shared__ float  sc[16 * S_];   // 32 KB score strip
    __shared__ half_t pp[16 * S_];   // 16 KB probabilities
    __shared__ float  red[16 * 8];   // reduction scratch
    __shared__ float  rowstat[16];

    int sb = blockIdx.x & 31;                 // query tile (S/16 = 32)
    int h  = (blockIdx.x >> 5) % H_;
    int b  = blockIdx.x / (32 * H_);
    int s0 = sb * 16;

    int wave = threadIdx.x >> 5;
    int lane = threadIdx.x & 31;
    int l = lane & 15, hi = lane >> 4;

    // chain ids for the 8 (row) elements this lane owns in C layout
    int cid_row[8];
#pragma unroll
    for (int r = 0; r < 8; ++r)
        cid_row[r] = chain[b * S_ + s0 + r + 8 * hi];

    size_t abase = ((size_t)(b * S_ + s0 + l)) * (H_ * D_) + h * D_;
    const half_t* aR = qr + abase;   // rope'd queries (A)
    const half_t* aN = qh + abase;   // plain queries  (A)

    // -------- phase 1: scores ------------------------------------------------
    for (int tt = wave; tt < S_ / 16; tt += 4) {
        int tcol = tt * 16 + l;
        size_t bbase = ((size_t)(b * S_ + tcol)) * (H_ * D_) + h * D_;
        const half_t* bR = kr + bbase;
        const half_t* bN = kh + bbase;

        v8f accR = {}, accN = {};
#pragma unroll
        for (int k0 = 0; k0 < D_; k0 += 32) {
            v16h aRf = load_a_frag(aR + k0, hi);
            v16h bRf = load_b_frag(bR + k0, hi);
            accR = __builtin_amdgcn_wmma_f32_16x16x32_f16(
                false, aRf, false, bRf, (short)0, accR, false, false);
            v16h aNf = load_a_frag(aN + k0, hi);
            v16h bNf = load_b_frag(bN + k0, hi);
            accN = __builtin_amdgcn_wmma_f32_16x16x32_f16(
                false, aNf, false, bNf, (short)0, accN, false, false);
        }

        int   cid_t = chain[b * S_ + tcol];
        float mterm = amask[b * S_ + tcol] ? 0.0f : -1e30f;
#pragma unroll
        for (int r = 0; r < 8; ++r) {
            float v = (cid_row[r] == cid_t ? accR[r] : accN[r]) * SCALE_;
            sc[(r + 8 * hi) * S_ + tcol] = v + mterm;
        }
    }
    __syncthreads();

    // -------- phase 2: softmax (8 threads per row, 64 cols each) -------------
    {
        int row = threadIdx.x >> 3;       // 0..15
        int chk = threadIdx.x & 7;        // 0..7
        int c0 = chk * 64;

        float m = -1e30f;
        for (int t = 0; t < 64; ++t) m = fmaxf(m, sc[row * S_ + c0 + t]);
        red[row * 8 + chk] = m;
        __syncthreads();
        if (chk == 0) {
            float mm = red[row * 8];
#pragma unroll
            for (int j = 1; j < 8; ++j) mm = fmaxf(mm, red[row * 8 + j]);
            rowstat[row] = mm;
        }
        __syncthreads();
        float mm = rowstat[row];

        float sum = 0.0f;
        for (int t = 0; t < 64; ++t) {
            float e = __expf(sc[row * S_ + c0 + t] - mm);
            sc[row * S_ + c0 + t] = e;
            sum += e;
        }
        red[row * 8 + chk] = sum;
        __syncthreads();
        if (chk == 0) {
            float ss = 0.0f;
#pragma unroll
            for (int j = 0; j < 8; ++j) ss += red[row * 8 + j];
            rowstat[row] = (ss > 0.0f) ? 1.0f / ss : 0.0f;   // nan_to_num
        }
        __syncthreads();
        float rinv = rowstat[row];
        for (int t = 0; t < 64; ++t)
            pp[row * S_ + c0 + t] = (half_t)(sc[row * S_ + c0 + t] * rinv);
    }
    __syncthreads();

    // -------- phase 3: P x V^T ----------------------------------------------
    int dd = wave * 16 + l;   // output feature column within head
    const half_t* vcol = vT + ((size_t)(b * H_ + h) * D_ + dd) * S_;

    v8f acc = {};
    for (int k0 = 0; k0 < S_; k0 += 32) {
        v16h af = load_a_frag(&pp[l * S_ + k0], hi);   // ds_read A-frag
        v16h bf = load_b_frag(vcol + k0, hi);
        acc = __builtin_amdgcn_wmma_f32_16x16x32_f16(
            false, af, false, bf, (short)0, acc, false, false);
    }

#pragma unroll
    for (int r = 0; r < 8; ++r) {
        size_t row = (size_t)b * S_ + s0 + r + 8 * hi;
        aout[row * (H_ * D_) + h * D_ + wave * 16 + l] = (half_t)acc[r];
    }
}

// ---------------------------------------------------------------------------
extern "C" void kernel_launch(void* const* d_in, const int* in_sizes, int n_in,
                              void* d_out, int out_size, void* d_ws, size_t ws_size,
                              hipStream_t stream) {
    const float* x  = (const float*)d_in[0];
    const int*   chain = (const int*)d_in[1];
    const unsigned char* amask = (const unsigned char*)d_in[2];
    const float* Wq = (const float*)d_in[3];
    const float* Wk = (const float*)d_in[4];
    const float* Wv = (const float*)d_in[5];
    const float* Wo = (const float*)d_in[6];
    float* out = (float*)d_out;

    const size_t NE = (size_t)ROWS_ * DM_;       // 6,291,456 elems
    const size_t HB = NE * sizeof(half_t);       // 12,582,912 B (256-aligned)
    const size_t WB = (size_t)DM_ * DM_ * sizeof(half_t);

    char* p = (char*)d_ws;
    half_t* xh  = (half_t*)p; p += HB;
    half_t* wqT = (half_t*)p; p += WB;
    half_t* wkT = (half_t*)p; p += WB;
    half_t* wvT = (half_t*)p; p += WB;
    half_t* woT = (half_t*)p; p += WB;
    half_t* qh  = (half_t*)p; p += HB;
    half_t* kh  = (half_t*)p; p += HB;
    half_t* vh  = (half_t*)p; p += HB;
    half_t* qrh = (half_t*)p; p += HB;
    half_t* krh = (half_t*)p; p += HB;
    half_t* vTp = (half_t*)p; p += HB;
    half_t* aoh = (half_t*)p; p += HB;
    (void)ws_size; (void)n_in; (void)in_sizes; (void)out_size;

    // stage 1: conversions
    cvt_f32_f16_kernel<<<4096, 256, 0, stream>>>(x, xh, (int)NE);
    transpose_cvt_kernel<<<2304, 256, 0, stream>>>(Wq, wqT, DM_, DM_);
    transpose_cvt_kernel<<<2304, 256, 0, stream>>>(Wk, wkT, DM_, DM_);
    transpose_cvt_kernel<<<2304, 256, 0, stream>>>(Wv, wvT, DM_, DM_);
    transpose_cvt_kernel<<<2304, 256, 0, stream>>>(Wo, woT, DM_, DM_);

    // stage 2: Q/K/V projections (8192x768 @ 768x768), f16 out
    // each wave: 16x64 tile -> (8192/16)*(768/64) = 6144 waves = 1536 blocks
    int gblk = (ROWS_ / 16) * (DM_ / 64) / 4;    // 1536
    gemm_tn_kernel<true><<<gblk, 128, 0, stream>>>(xh, wqT, qh, ROWS_, DM_, DM_);
    gemm_tn_kernel<true><<<gblk, 128, 0, stream>>>(xh, wkT, kh, ROWS_, DM_, DM_);
    gemm_tn_kernel<true><<<gblk, 128, 0, stream>>>(xh, wvT, vh, ROWS_, DM_, DM_);

    // stage 3: RoPE + V transpose
    rope_vt_kernel<<<12288, 256, 0, stream>>>(qh, kh, vh, qrh, krh, vTp);

    // stage 4: chain-aware attention
    attn_kernel<<<B_ * H_ * (S_ / 16), 128, 0, stream>>>(
        qh, kh, qrh, krh, vTp, chain, amask, aoh);

    // stage 5: output projection, f32 out -> d_out
    gemm_tn_kernel<false><<<gblk, 128, 0, stream>>>(aoh, woT, out, ROWS_, DM_, DM_);
}
